// my_LSTM_35785667510500
// MI455X (gfx1250) — matrix operand order repaired and checked by reading, not compiled, not run on hardware
//
#include <hip/hip_runtime.h>

// ---------------------------------------------------------------------------
// LSTM on MI455X (gfx1250, wave32, WMMA bf16->f32)
//   B=64, S=512, D=H=F=1024
// Strategy:
//   1) convert x to bf16
//   2) pack W_g[:D], W_g[D:], W_out into WMMA B-fragment layout (bf16)
//   3) one big GEMM: Gx = x @ W_gx + b_g            (parallel, 275 GFLOP)
//   4) 512 sequential small GEMM+elementwise steps: g = Gx_t + h @ W_gh
//   5) one big GEMM: out = Hall @ W_out + b_out     (parallel, 69 GFLOP)
// WMMA K-loops are software-pipelined; __builtin_amdgcn_sched_barrier(0)
// fences stop the scheduler from sinking prefetch loads back under the
// WMMAs (which forced s_wait_loadcnt 0 before every matrix op).
// ---------------------------------------------------------------------------

typedef __attribute__((ext_vector_type(16))) __bf16 v16bf;
typedef __attribute__((ext_vector_type(8)))  __bf16 v8bf;
typedef __attribute__((ext_vector_type(8)))  float  v8f;

constexpr int Bc = 64, Sc = 512, Dc = 1024, Hc = 1024, Fc = 1024;
constexpr int G4 = 4 * Hc;           // 4096 gate columns
constexpr int KT_D = Dc / 32;        // 32 K-tiles for D
constexpr int KT_H = Hc / 32;        // 32 K-tiles for H

#define DEVFN static __device__ __forceinline__

#if defined(__has_builtin) && __has_builtin(__builtin_amdgcn_sched_barrier)
#define SCHED_FENCE() __builtin_amdgcn_sched_barrier(0)
#else
#define SCHED_FENCE() __asm__ volatile("" ::: "memory")
#endif

// ---- A fragment loader: row-major bf16, lane L holds row (L&15).
// Contiguous K order (matches packed B): lanes0-15 K=0..15, lanes16-31 K=16..31
// -> one 32B load at byte offset 32*hi.
DEVFN v16bf load_a_frag(const __bf16* rowk, int hi) {
  return *(const v16bf*)(rowk + hi * 16);
}

DEVFN v8f wmma_bf16(v16bf a, v16bf b, v8f c) {
  return __builtin_amdgcn_wmma_f32_16x16x32_bf16(false, a, false, b,
                                                 (short)0, c, false, false);
}

DEVFN float sigmoid_f(float x) { return 1.0f / (1.0f + __expf(-x)); }
DEVFN float tanh_f(float x) {
  float e = __expf(2.0f * x);
  return (e - 1.0f) / (e + 1.0f);
}

// ---------------------------------------------------------------------------
__global__ void k_cvt_bf16(const float* __restrict__ src,
                           __bf16* __restrict__ dst, long long n) {
  long long i = (long long)blockIdx.x * blockDim.x + threadIdx.x;
  long long stride = (long long)gridDim.x * blockDim.x;
  for (; i < n; i += stride) dst[i] = (__bf16)src[i];
}

// Pack gate weights [D+H, H] x4 -> B-fragment layout for columns [4H].
// Packed B (this code's contiguous K order): lane L holds column N=L&15; lanes0-15 rows K=0..15,
// lanes16-31 rows K=16..31 (2 per VGPR). Fragment (nt,kt) stored contiguously
// at ((nt*KT + kt)*512 + lane*16 + e).
__global__ void k_pack_gateW(const float* __restrict__ Wf,
                             const float* __restrict__ Wi,
                             const float* __restrict__ Wc,
                             const float* __restrict__ Wo,
                             __bf16* __restrict__ dst, int k_base) {
  int t = blockIdx.x * blockDim.x + threadIdx.x;
  int lane = t & 31, fid = t >> 5;
  int nt = fid / KT_D, kt = fid % KT_D;
  if (nt >= G4 / 16) return;
  int hi = lane >> 4;
  int n = nt * 16 + (lane & 15);
  const float* W = (n < Hc) ? Wf : (n < 2 * Hc) ? Wi : (n < 3 * Hc) ? Wc : Wo;
  int c = n & (Hc - 1);
#pragma unroll
  for (int e = 0; e < 16; ++e) {
    int K = hi * 16 + e;
    dst[(size_t)fid * 512 + lane * 16 + e] =
        (__bf16)W[(size_t)(k_base + kt * 32 + K) * Hc + c];
  }
}

// Pack a single row-major [Kdim, Ndim] weight into B fragments.
__global__ void k_pack_W(const float* __restrict__ W, __bf16* __restrict__ dst,
                         int Kdim, int Ndim) {
  int t = blockIdx.x * blockDim.x + threadIdx.x;
  int lane = t & 31, fid = t >> 5;
  int KT = Kdim / 32;
  int nt = fid / KT, kt = fid % KT;
  if (nt >= Ndim / 16) return;
  int hi = lane >> 4;
  int n = nt * 16 + (lane & 15);
#pragma unroll
  for (int e = 0; e < 16; ++e) {
    int K = hi * 16 + e;
    dst[(size_t)fid * 512 + lane * 16 + e] =
        (__bf16)W[(size_t)(kt * 32 + K) * Ndim + n];
  }
}

__global__ void k_bias_concat(const float* __restrict__ bf_,
                              const float* __restrict__ bi_,
                              const float* __restrict__ bc_,
                              const float* __restrict__ bo_,
                              float* __restrict__ bg) {
  int i = blockIdx.x * blockDim.x + threadIdx.x;
  if (i >= G4) return;
  const float* b = (i < Hc) ? bf_ : (i < 2 * Hc) ? bi_ : (i < 3 * Hc) ? bc_ : bo_;
  bg[i] = b[i & (Hc - 1)];
}

__global__ void k_init_state(const float* __restrict__ h0,
                             const float* __restrict__ c0,
                             __bf16* __restrict__ Hall, float* __restrict__ cst) {
  int i = blockIdx.x * blockDim.x + threadIdx.x;
  if (i >= Bc * Hc) return;
  Hall[i] = (__bf16)h0[i];  // block 0 of Hall = h_{-1}
  cst[i] = c0[i];
}

// ---------------------------------------------------------------------------
// Gx = xb @ Wgx_pack + b_g   ->  bf16 [B*S, 4H]   (rows ordered b*S+s)
// Block: 256 thr (8 waves, 2M x 4N). Wave tile: 16M x 64N (4 accumulators).
__global__ void __launch_bounds__(256)
k_gemm_gx(const __bf16* __restrict__ A, const __bf16* __restrict__ Bp,
          const float* __restrict__ bg, __bf16* __restrict__ Gx) {
  const int tid = threadIdx.x, lane = tid & 31, wid = tid >> 5;
  const int m = lane & 15, hi = lane >> 4;
  const int nblk = blockIdx.x & 15, mblk = blockIdx.x >> 4;   // 16 N-blocks of 256
  const int rowA = mblk * 32 + (wid >> 2) * 16 + m;
  const int n0 = nblk * 256 + (wid & 3) * 64;
  const __bf16* arow = A + (size_t)rowA * Dc;
  const __bf16* bbase = Bp + ((size_t)(n0 >> 4) * KT_D) * 512 + lane * 16;

  v8f acc[4] = {};
  // prime double buffer with kt=0
  v16bf aC = load_a_frag(arow, hi);
  v16bf bC0 = *(const v16bf*)(bbase);
  v16bf bC1 = *(const v16bf*)(bbase + (size_t)1 * KT_D * 512);
  v16bf bC2 = *(const v16bf*)(bbase + (size_t)2 * KT_D * 512);
  v16bf bC3 = *(const v16bf*)(bbase + (size_t)3 * KT_D * 512);
#pragma unroll 4
  for (int kt = 0; kt < KT_D; ++kt) {
    const int ktn = (kt + 1) & (KT_D - 1);   // wraps: last prefetch is a no-op reload
    v16bf aN = load_a_frag(arow + ktn * 32, hi);
    const __bf16* bpn = bbase + (size_t)ktn * 512;
    v16bf bN0 = *(const v16bf*)(bpn);
    v16bf bN1 = *(const v16bf*)(bpn + (size_t)1 * KT_D * 512);
    v16bf bN2 = *(const v16bf*)(bpn + (size_t)2 * KT_D * 512);
    v16bf bN3 = *(const v16bf*)(bpn + (size_t)3 * KT_D * 512);
    SCHED_FENCE();   // keep prefetch loads above the matrix ops
    acc[0] = wmma_bf16(aC, bC0, acc[0]);
    acc[1] = wmma_bf16(aC, bC1, acc[1]);
    acc[2] = wmma_bf16(aC, bC2, acc[2]);
    acc[3] = wmma_bf16(aC, bC3, acc[3]);
    SCHED_FENCE();   // keep next iteration's work below
    aC = aN; bC0 = bN0; bC1 = bN1; bC2 = bN2; bC3 = bN3;
  }
  const int r0 = mblk * 32 + (wid >> 2) * 16 + hi * 8;  // C rows j+8*hi
#pragma unroll
  for (int t = 0; t < 4; ++t) {
    int col = n0 + t * 16 + m;
    float bb = bg[col];
#pragma unroll
    for (int j = 0; j < 8; ++j)
      Gx[(size_t)(r0 + j) * G4 + col] = (__bf16)(acc[t][j] + bb);
  }
}

// ---------------------------------------------------------------------------
// One recurrence step: g = Gx[:,s,:] + h_prev @ Wgh; update c,h; h->Hall.
// 256 waves total: blockDim 128 (4 waves) x 64 blocks.
// Each wave: one 16-batch x 16-col slice across ALL FOUR gates (4 accs),
// so the elementwise LSTM update is local to the wave's fragments.
__global__ void __launch_bounds__(128)
k_step(__bf16* __restrict__ Hall, const __bf16* __restrict__ Wgh,
       const __bf16* __restrict__ Gx, float* __restrict__ cst, int s) {
  const int tid = threadIdx.x, lane = tid & 31, wid = tid >> 5;
  const int wg = blockIdx.x * 4 + wid;     // 0..255
  const int mt = wg >> 6;                  // 0..3  (batch tile)
  const int n0 = (wg & 63) * 16;           // hidden-col tile
  const int m = lane & 15, hi = lane >> 4;

  const __bf16* hprev = Hall + (size_t)s * Bc * Hc;
  const __bf16* arow = hprev + (size_t)(mt * 16 + m) * Hc;
  // per-gate fragment base pointers (gate q columns start at q*Hc)
  const __bf16* wb0 = Wgh + ((size_t)((0 * Hc + n0) >> 4) * KT_H) * 512 + lane * 16;
  const __bf16* wb1 = Wgh + ((size_t)((1 * Hc + n0) >> 4) * KT_H) * 512 + lane * 16;
  const __bf16* wb2 = Wgh + ((size_t)((2 * Hc + n0) >> 4) * KT_H) * 512 + lane * 16;
  const __bf16* wb3 = Wgh + ((size_t)((3 * Hc + n0) >> 4) * KT_H) * 512 + lane * 16;

  v8f acc[4] = {};
  v16bf aC = load_a_frag(arow, hi);
  v16bf bC0 = *(const v16bf*)(wb0);
  v16bf bC1 = *(const v16bf*)(wb1);
  v16bf bC2 = *(const v16bf*)(wb2);
  v16bf bC3 = *(const v16bf*)(wb3);
#pragma unroll 4
  for (int kt = 0; kt < KT_H; ++kt) {
    const int ktn = (kt + 1) & (KT_H - 1);
    v16bf aN = load_a_frag(arow + ktn * 32, hi);
    v16bf bN0 = *(const v16bf*)(wb0 + (size_t)ktn * 512);
    v16bf bN1 = *(const v16bf*)(wb1 + (size_t)ktn * 512);
    v16bf bN2 = *(const v16bf*)(wb2 + (size_t)ktn * 512);
    v16bf bN3 = *(const v16bf*)(wb3 + (size_t)ktn * 512);
    SCHED_FENCE();
    acc[0] = wmma_bf16(aC, bC0, acc[0]);
    acc[1] = wmma_bf16(aC, bC1, acc[1]);
    acc[2] = wmma_bf16(aC, bC2, acc[2]);
    acc[3] = wmma_bf16(aC, bC3, acc[3]);
    SCHED_FENCE();
    aC = aN; bC0 = bN0; bC1 = bN1; bC2 = bN2; bC3 = bN3;
  }

  const int col = n0 + m;
  __bf16* hout = Hall + (size_t)(s + 1) * Bc * Hc;
#pragma unroll
  for (int j = 0; j < 8; ++j) {
    int batch = mt * 16 + hi * 8 + j;
    const __bf16* gxr = Gx + ((size_t)batch * Sc + s) * G4;  // bias baked in
    float gf = acc[0][j] + (float)gxr[0 * Hc + col];
    float gi = acc[1][j] + (float)gxr[1 * Hc + col];
    float gc = acc[2][j] + (float)gxr[2 * Hc + col];
    float go = acc[3][j] + (float)gxr[3 * Hc + col];
    float f_ = sigmoid_f(gf);
    float i_ = sigmoid_f(gi);
    float ch = tanh_f(gc);
    float o_ = sigmoid_f(go);
    size_t ci = (size_t)batch * Hc + col;
    float cn = f_ * cst[ci] + i_ * ch;
    cst[ci] = cn;
    hout[ci] = (__bf16)(o_ * tanh_f(cn));
  }
}

// ---------------------------------------------------------------------------
// out[b,s,:] = Hall[s*B+b,:] @ Wout_pack + b_out   (f32 output, row remap)
__global__ void __launch_bounds__(256)
k_gemm_out(const __bf16* __restrict__ A, const __bf16* __restrict__ Bp,
           const float* __restrict__ bout, float* __restrict__ out) {
  const int tid = threadIdx.x, lane = tid & 31, wid = tid >> 5;
  const int m = lane & 15, hi = lane >> 4;
  const int nblk = blockIdx.x & 3, mblk = blockIdx.x >> 2;  // 4 N-blocks of 256
  const int rowA = mblk * 32 + (wid >> 2) * 16 + m;
  const int n0 = nblk * 256 + (wid & 3) * 64;
  const __bf16* arow = A + (size_t)rowA * Hc;
  const __bf16* bbase = Bp + ((size_t)(n0 >> 4) * KT_H) * 512 + lane * 16;

  v8f acc[4] = {};
  v16bf aC = load_a_frag(arow, hi);
  v16bf bC0 = *(const v16bf*)(bbase);
  v16bf bC1 = *(const v16bf*)(bbase + (size_t)1 * KT_H * 512);
  v16bf bC2 = *(const v16bf*)(bbase + (size_t)2 * KT_H * 512);
  v16bf bC3 = *(const v16bf*)(bbase + (size_t)3 * KT_H * 512);
#pragma unroll 4
  for (int kt = 0; kt < KT_H; ++kt) {
    const int ktn = (kt + 1) & (KT_H - 1);
    v16bf aN = load_a_frag(arow + ktn * 32, hi);
    const __bf16* bpn = bbase + (size_t)ktn * 512;
    v16bf bN0 = *(const v16bf*)(bpn);
    v16bf bN1 = *(const v16bf*)(bpn + (size_t)1 * KT_H * 512);
    v16bf bN2 = *(const v16bf*)(bpn + (size_t)2 * KT_H * 512);
    v16bf bN3 = *(const v16bf*)(bpn + (size_t)3 * KT_H * 512);
    SCHED_FENCE();
    acc[0] = wmma_bf16(aC, bC0, acc[0]);
    acc[1] = wmma_bf16(aC, bC1, acc[1]);
    acc[2] = wmma_bf16(aC, bC2, acc[2]);
    acc[3] = wmma_bf16(aC, bC3, acc[3]);
    SCHED_FENCE();
    aC = aN; bC0 = bN0; bC1 = bN1; bC2 = bN2; bC3 = bN3;
  }
  const int r0 = mblk * 32 + (wid >> 2) * 16 + hi * 8;
#pragma unroll
  for (int t = 0; t < 4; ++t) {
    int col = n0 + t * 16 + m;
    float bb = bout[col];
#pragma unroll
    for (int j = 0; j < 8; ++j) {
      int ar = r0 + j;               // A row = s*B + b
      int s_ = ar >> 6;              // /64
      int b_ = ar & 63;
      out[((size_t)b_ * Sc + s_) * Fc + col] = acc[t][j] + bb;
    }
  }
}

// ---------------------------------------------------------------------------
extern "C" void kernel_launch(void* const* d_in, const int* in_sizes, int n_in,
                              void* d_out, int out_size, void* d_ws,
                              size_t ws_size, hipStream_t stream) {
  const float* x    = (const float*)d_in[0];
  const float* h0   = (const float*)d_in[1];
  const float* c0   = (const float*)d_in[2];
  const float* Wf   = (const float*)d_in[3];
  const float* bf_  = (const float*)d_in[4];
  const float* Wi   = (const float*)d_in[5];
  const float* bi_  = (const float*)d_in[6];
  const float* Wc   = (const float*)d_in[7];
  const float* bc_  = (const float*)d_in[8];
  const float* Wo   = (const float*)d_in[9];
  const float* bo_  = (const float*)d_in[10];
  const float* Wout = (const float*)d_in[11];
  const float* bout = (const float*)d_in[12];
  float* out = (float*)d_out;
  (void)in_sizes; (void)n_in; (void)out_size; (void)ws_size;

  char* base = (char*)d_ws;
  size_t off = 0;
  auto take = [&](size_t bytes) -> char* {
    char* p = base + off;
    off = (off + bytes + 255) & ~(size_t)255;
    return p;
  };
  __bf16* xb   = (__bf16*)take((size_t)Bc * Sc * Dc * 2);        //  64 MB
  __bf16* Wgx  = (__bf16*)take((size_t)Dc * G4 * 2);             //   8 MB
  __bf16* Wgh  = (__bf16*)take((size_t)Hc * G4 * 2);             //   8 MB
  __bf16* Wop  = (__bf16*)take((size_t)Hc * Fc * 2);             //   2 MB
  float*  bg   = (float*) take((size_t)G4 * 4);                  //  16 KB
  __bf16* Gx   = (__bf16*)take((size_t)Bc * Sc * G4 * 2);        // 256 MB
  __bf16* Hall = (__bf16*)take((size_t)(Sc + 1) * Bc * Hc * 2);  //  64 MB
  float*  cst  = (float*) take((size_t)Bc * Hc * 4);             // 256 KB

  // --- setup / packing -----------------------------------------------------
  k_cvt_bf16<<<32768, 256, 0, stream>>>(x, xb, (long long)Bc * Sc * Dc);
  k_pack_gateW<<<1024, 256, 0, stream>>>(Wf, Wi, Wc, Wo, Wgx, 0);    // x-part
  k_pack_gateW<<<1024, 256, 0, stream>>>(Wf, Wi, Wc, Wo, Wgh, Dc);   // h-part
  k_pack_W<<<256, 256, 0, stream>>>(Wout, Wop, Hc, Fc);
  k_bias_concat<<<16, 256, 0, stream>>>(bf_, bi_, bc_, bo_, bg);
  k_init_state<<<256, 256, 0, stream>>>(h0, c0, Hall, cst);

  // --- batched x-part of gates (bias folded in) ----------------------------
  k_gemm_gx<<<16384, 256, 0, stream>>>(xb, Wgx, bg, Gx);

  // --- 512 sequential recurrence steps (Wgh stays L2-resident) -------------
  for (int s = 0; s < Sc; ++s)
    k_step<<<64, 128, 0, stream>>>(Hall, Wgh, Gx, cst, s);

  // --- batched output projection -------------------------------------------
  k_gemm_out<<<4096, 256, 0, stream>>>(Hall + (size_t)Bc * Hc, Wop, bout, out);
}
